// GATv2_274877907718
// MI455X (gfx1250) — matrix operand order repaired and checked by reading, not compiled, not run on hardware
//
#include <hip/hip_runtime.h>
#include <hip/hip_bf16.h>
#include <math.h>

// ---------------------------------------------------------------------------
// GATv2 3-layer pipeline for gfx1250 (MI455X).
//  - Dense projections: pre-packed bf16 hi/lo fragments + v_wmma_f32_16x16x32_bf16
//    (3-product split => ~fp32 accuracy), B tile staged in LDS (async if available)
//  - Edge softmax: ordered-uint atomicMax + atomicAdd f32, float4-vectorized
// ---------------------------------------------------------------------------

typedef __attribute__((ext_vector_type(16))) __bf16 v16bf;
typedef __attribute__((ext_vector_type(8)))  float  v8f;
typedef int v4i __attribute__((vector_size(16)));

#define NEG_SLOPE 0.2f
#define KSTEPS 4          // all GEMMs have K = 128 = 4 * 32

#define AS1 __attribute__((address_space(1)))
#define AS3 __attribute__((address_space(3)))

#if defined(__HIP_DEVICE_COMPILE__) && \
    __has_builtin(__builtin_amdgcn_global_load_async_to_lds_b128) && \
    __has_builtin(__builtin_amdgcn_s_wait_asynccnt)
#define USE_ASYNC_LDS 1
#else
#define USE_ASYNC_LDS 0
#endif

// ------------------------- fragment pre-pack --------------------------------
// A fragment layout (16-bit A 16x32, ISA table): lane l: row = l&15,
// half = l>>4; chunk element j(0..15): k = ks*32 + half*8 + (j<8 ? j : j+8).
// Chunk of 32 bf16 per (tile,ks,lane): [0..15]=hi, [16..31]=lo.
__global__ void pack_a_kernel(const float* __restrict__ A,
                              __bf16* __restrict__ out, int M, int K)
{
    long idx = (long)blockIdx.x * blockDim.x + threadIdx.x;
    const int ksteps = K / 32;
    long total = (long)(M / 16) * ksteps * 32 * 16;
    if (idx >= total) return;
    const int j    = (int)(idx & 15);
    const int lane = (int)((idx >> 4) & 31);
    long rest      = idx >> 9;
    const int ks   = (int)(rest % ksteps);
    const int rt   = (int)(rest / ksteps);
    const int row  = rt * 16 + (lane & 15);
    const int half = lane >> 4;
    const int k    = ks * 32 + half * 8 + (j < 8 ? j : j + 8);
    const float x  = A[(size_t)row * K + k];
    const __bf16 hi = (__bf16)x;
    const __bf16 lo = (__bf16)(x - (float)hi);
    const size_t base = (((size_t)rt * ksteps + ks) * 32 + lane) * 32;
    out[base + j]      = hi;
    out[base + 16 + j] = lo;
}

// B fragment layout (16-bit B 32x16): lane l: col = l&15, half = l>>4;
// chunk element j(0..15): k = ks*32 + half*8 + (j<8 ? j : j+8).
__global__ void pack_b_kernel(const float* __restrict__ B,
                              __bf16* __restrict__ out, int K, int Nout)
{
    long idx = (long)blockIdx.x * blockDim.x + threadIdx.x;
    const int ksteps = K / 32;
    long total = (long)(Nout / 16) * ksteps * 32 * 16;
    if (idx >= total) return;
    const int j    = (int)(idx & 15);
    const int lane = (int)((idx >> 4) & 31);
    long rest      = idx >> 9;
    const int ks   = (int)(rest % ksteps);
    const int ct   = (int)(rest / ksteps);
    const int col  = ct * 16 + (lane & 15);
    const int half = lane >> 4;
    const int k    = ks * 32 + half * 8 + (j < 8 ? j : j + 8);
    const float x  = B[(size_t)k * Nout + col];
    const __bf16 hi = (__bf16)x;
    const __bf16 lo = (__bf16)(x - (float)hi);
    const size_t base = (((size_t)ct * ksteps + ks) * 32 + lane) * 32;
    out[base + j]      = hi;
    out[base + 16 + j] = lo;
}

// ------------------------- WMMA GEMM on packed fragments --------------------
// C[M,Nout] = A[M,128] * B[128,Nout]; one 16x16 tile per wave, 8 waves/block.
// B tile (8KB) staged in LDS once per block (async global->LDS when available).
__global__ __launch_bounds__(256)
void gemm_wmma_packed(const __bf16* __restrict__ pA,
                      const __bf16* __restrict__ pB,
                      float* __restrict__ C,
                      int M, int Nout)
{
    __shared__ __align__(16) __bf16 sB[KSTEPS * 32 * 32];   // 8 KB

    const int tid  = threadIdx.x;
    const int lane = tid & 31;
    const int wave = tid >> 5;

    // ---- stage this block's packed-B column tile into LDS
    const __bf16* gB = pB + (size_t)blockIdx.y * (KSTEPS * 32 * 32);
#if USE_ASYNC_LDS
    {
        const AS1 char* g = (const AS1 char*)(const char*)gB;
        AS3 char*       l = (AS3 char*)sB;
        for (int i = tid; i < (KSTEPS * 32 * 32 * 2) / 16; i += 256)
            __builtin_amdgcn_global_load_async_to_lds_b128(
                (AS1 v4i*)(g + 16 * i), (AS3 v4i*)(l + 16 * i), 0, 0);
        __builtin_amdgcn_s_wait_asynccnt(0);
    }
    __syncthreads();
#else
    {
        const uint4* g4 = (const uint4*)gB;
        uint4*       s4 = (uint4*)sB;
        for (int i = tid; i < (KSTEPS * 32 * 32 * 2) / 16; i += 256)
            s4[i] = g4[i];
    }
    __syncthreads();
#endif

    const int rowTile = blockIdx.x * 8 + wave;
    if (rowTile * 16 >= M) return;               // wave-uniform; after barrier

    const int half = lane >> 4;
    v8f acc = {};

    #pragma unroll
    for (int ks = 0; ks < KSTEPS; ++ks) {
        const v16bf* av = (const v16bf*)(pA +
            (((size_t)rowTile * KSTEPS + ks) * 32 + lane) * 32);
        if (ks + 1 < KSTEPS)
            __builtin_prefetch((const __bf16*)av + 32 * 32, 0, 1);
        const v16bf ahi = av[0];
        const v16bf alo = av[1];
        const v16bf* bv = (const v16bf*)(sB + ((size_t)ks * 32 + lane) * 32);
        const v16bf bhi = bv[0];
        const v16bf blo = bv[1];
        acc = __builtin_amdgcn_wmma_f32_16x16x32_bf16(false, ahi, false, bhi,
                                                      (short)0, acc, false, false);
        acc = __builtin_amdgcn_wmma_f32_16x16x32_bf16(false, alo, false, bhi,
                                                      (short)0, acc, false, false);
        acc = __builtin_amdgcn_wmma_f32_16x16x32_bf16(false, ahi, false, blo,
                                                      (short)0, acc, false, false);
    }

    const int col = blockIdx.y * 16 + (lane & 15);
    #pragma unroll
    for (int v = 0; v < 8; ++v) {
        const int m = rowTile * 16 + v + 8 * half;
        C[(size_t)m * Nout + col] = acc[v];
    }
}

// ------------------------- float <-> ordered-uint (for atomic max) ----------
__device__ __forceinline__ unsigned f2ord(float f) {
    unsigned u = __float_as_uint(f);
    return (u & 0x80000000u) ? ~u : (u | 0x80000000u);
}
__device__ __forceinline__ float ord2f(unsigned e) {
    unsigned u = (e & 0x80000000u) ? (e & 0x7FFFFFFFu) : ~e;
    return __uint_as_float(u);
}

// ------------------------- fill ---------------------------------------------
__global__ void fill_u32(unsigned* __restrict__ p, unsigned v, long n) {
    long i = (long)blockIdx.x * blockDim.x + threadIdx.x;
    if (i < n) p[i] = v;
}

// ------------------------- edge phase ---------------------------------------
__global__ void edge_logits_kernel(const float* __restrict__ feat,
                                   const int* __restrict__ src,
                                   const int* __restrict__ dst,
                                   const float* __restrict__ attn,
                                   float* __restrict__ logits,
                                   unsigned* __restrict__ nodemax,
                                   int E, int H, int D)
{
    long idx = (long)blockIdx.x * blockDim.x + threadIdx.x;
    if (idx >= (long)E * H) return;
    const int e = (int)(idx / H);
    const int h = (int)(idx % H);
    const int s = src[e], d = dst[e];
    const int F = H * D;
    const float4* fs = (const float4*)(feat + (size_t)s * F + h * D);
    const float4* fd = (const float4*)(feat + (size_t)d * F + h * D);
    const float4* at = (const float4*)(attn + h * D);
    float acc = 0.f;
    for (int j = 0; j < D / 4; ++j) {
        const float4 a = fs[j], b = fd[j], w = at[j];
        float x;
        x = a.x + b.x; x = (x > 0.f) ? x : NEG_SLOPE * x; acc += x * w.x;
        x = a.y + b.y; x = (x > 0.f) ? x : NEG_SLOPE * x; acc += x * w.y;
        x = a.z + b.z; x = (x > 0.f) ? x : NEG_SLOPE * x; acc += x * w.z;
        x = a.w + b.w; x = (x > 0.f) ? x : NEG_SLOPE * x; acc += x * w.w;
    }
    logits[idx] = acc;
    atomicMax(&nodemax[(size_t)d * H + h], f2ord(acc));
}

__global__ void edge_exp_kernel(float* __restrict__ logits,
                                const int* __restrict__ dst,
                                const unsigned* __restrict__ nodemax,
                                float* __restrict__ denom,
                                int E, int H)
{
    long idx = (long)blockIdx.x * blockDim.x + threadIdx.x;
    if (idx >= (long)E * H) return;
    const int e = (int)(idx / H);
    const int h = (int)(idx % H);
    const int d = dst[e];
    const float m  = ord2f(nodemax[(size_t)d * H + h]);
    const float ex = expf(logits[idx] - m);
    logits[idx] = ex;
    atomicAdd(&denom[(size_t)d * H + h], ex);
}

// rst[dst, 4f..4f+3] += (ex/denom) * feat[src, 4f..4f+3]
__global__ void edge_agg_kernel(const float* __restrict__ feat,
                                const int* __restrict__ src,
                                const int* __restrict__ dst,
                                const float* __restrict__ ex,
                                const float* __restrict__ denom,
                                float* __restrict__ rst,
                                int E, int H, int D)
{
    const int F  = H * D;
    const int F4 = F / 4;
    long idx = (long)blockIdx.x * blockDim.x + threadIdx.x;
    if (idx >= (long)E * F4) return;
    const int e  = (int)(idx / F4);
    const int f4 = (int)(idx % F4);
    const int h  = (4 * f4) / D;
    const int s = src[e], d = dst[e];
    const float a = ex[(size_t)e * H + h] / denom[(size_t)d * H + h];
    const float4 v = *(const float4*)(feat + (size_t)s * F + 4 * f4);
    float* r = rst + (size_t)d * F + 4 * f4;
    atomicAdd(r + 0, a * v.x);
    atomicAdd(r + 1, a * v.y);
    atomicAdd(r + 2, a * v.z);
    atomicAdd(r + 3, a * v.w);
}

__global__ void elu_residual_kernel(float* __restrict__ rst,
                                    const float* __restrict__ res,
                                    long n)
{
    long i = (long)blockIdx.x * blockDim.x + threadIdx.x;
    if (i >= n) return;
    float v = rst[i] + (res ? res[i] : 0.f);
    rst[i] = (v > 0.f) ? v : expm1f(v);
}

__global__ void final_mean_kernel(const float* __restrict__ rst,
                                  const float* __restrict__ res,
                                  float* __restrict__ out,
                                  int N, int HO, int C)
{
    long idx = (long)blockIdx.x * blockDim.x + threadIdx.x;
    if (idx >= (long)N * C) return;
    const int n = (int)(idx / C);
    const int c = (int)(idx % C);
    float s = 0.f;
    for (int ho = 0; ho < HO; ++ho) {
        const size_t k = (size_t)n * HO * C + (size_t)ho * C + c;
        s += rst[k] + res[k];
    }
    out[idx] = s / (float)HO;
}

// ---------------------------------------------------------------------------
static inline long cdiv(long a, long b) { return (a + b - 1) / b; }

static void run_gemm(const float* A, const float* B, float* C,
                     __bf16* pA, __bf16* pB, bool packA,
                     int M, int K, int Nout, hipStream_t stream)
{
    const long TB = 256;
    if (packA) {
        long ta = (long)(M / 16) * (K / 32) * 32 * 16;
        pack_a_kernel<<<dim3(cdiv(ta, TB)), dim3(TB), 0, stream>>>(A, pA, M, K);
    }
    long tb = (long)(Nout / 16) * (K / 32) * 32 * 16;
    pack_b_kernel<<<dim3(cdiv(tb, TB)), dim3(TB), 0, stream>>>(B, pB, K, Nout);
    const int rowTiles = (M + 15) / 16;
    gemm_wmma_packed<<<dim3((rowTiles + 7) / 8, Nout / 16), dim3(256), 0, stream>>>(
        pA, pB, C, M, Nout);
}

static void run_edge_phase(const float* feat, const int* src, const int* dst,
                           const float* attn, float* logits, unsigned* nodemax,
                           float* denom, float* rst,
                           int N, int E, int H, int D, hipStream_t stream)
{
    const int F = H * D;
    const long TB = 256;
    fill_u32<<<dim3(cdiv((long)N * H, TB)), dim3(TB), 0, stream>>>(nodemax, 0u, (long)N * H);
    fill_u32<<<dim3(cdiv((long)N * H, TB)), dim3(TB), 0, stream>>>((unsigned*)denom, 0u, (long)N * H);
    fill_u32<<<dim3(cdiv((long)N * F, TB)), dim3(TB), 0, stream>>>((unsigned*)rst, 0u, (long)N * F);

    edge_logits_kernel<<<dim3(cdiv((long)E * H, TB)), dim3(TB), 0, stream>>>(
        feat, src, dst, attn, logits, nodemax, E, H, D);
    edge_exp_kernel<<<dim3(cdiv((long)E * H, TB)), dim3(TB), 0, stream>>>(
        logits, dst, nodemax, denom, E, H);
    edge_agg_kernel<<<dim3(cdiv((long)E * (F / 4), TB)), dim3(TB), 0, stream>>>(
        feat, src, dst, logits, denom, rst, E, H, D);
}

extern "C" void kernel_launch(void* const* d_in, const int* in_sizes, int n_in,
                              void* d_out, int out_size, void* d_ws, size_t ws_size,
                              hipStream_t stream)
{
    const float* inputs = (const float*)d_in[0];   // [N,128]
    const int*   src    = (const int*)  d_in[1];   // [E]
    const int*   dst    = (const int*)  d_in[2];   // [E]
    const float* W0     = (const float*)d_in[3];   // [128,128]
    const float* attn0  = (const float*)d_in[4];   // [4,32]
    const float* W1     = (const float*)d_in[5];   // [128,128]
    const float* attn1  = (const float*)d_in[6];   // [4,32]
    const float* W2     = (const float*)d_in[7];   // [128,240]
    const float* attn2  = (const float*)d_in[8];   // [6,40]
    const float* Wres2  = (const float*)d_in[9];   // [128,240]

    const int F_IN = 128;
    const int N = in_sizes[0] / F_IN;
    const int E = in_sizes[1];
    const int H = 4, D = 32, F01 = 128;
    const int HO = 6, C = 40, F2 = 240;

    // ---- workspace layout (fp32 element units; pieces back to back, no padding) ----
    float* ws = (float*)d_ws;
    size_t o = 0;
    float*    feat  = ws + o; o += (size_t)N * F2;
    float*    rstA  = ws + o; o += (size_t)N * F2;
    float*    rstB  = ws + o; o += (size_t)N * F2;
    float*    res2  = ws + o; o += (size_t)N * F2;
    float*    logit = ws + o; o += (size_t)E * HO;
    unsigned* nmax  = (unsigned*)(ws + o); o += (size_t)N * HO;
    float*    denom = ws + o; o += (size_t)N * HO;
    __bf16*   packA = (__bf16*)(ws + o); o += (size_t)N * F_IN;      // 2*N*128 bf16
    __bf16*   packB = (__bf16*)(ws + o); o += 32768;                 // 2*128*240 bf16 max
    (void)ws_size; (void)n_in; (void)out_size;

    // ===================== layer 0 =====================
    run_gemm(inputs, W0, feat, packA, packB, true, N, F_IN, F01, stream);
    run_edge_phase(feat, src, dst, attn0, logit, nmax, denom, rstA, N, E, H, D, stream);
    elu_residual_kernel<<<dim3(cdiv((long)N * F01, 256)), dim3(256), 0, stream>>>(
        rstA, nullptr, (long)N * F01);

    // ===================== layer 1 (identity residual) =====================
    run_gemm(rstA, W1, feat, packA, packB, true, N, F01, F01, stream);
    run_edge_phase(feat, src, dst, attn1, logit, nmax, denom, rstB, N, E, H, D, stream);
    elu_residual_kernel<<<dim3(cdiv((long)N * F01, 256)), dim3(256), 0, stream>>>(
        rstB, rstA, (long)N * F01);

    // ===================== layer 2 (linear residual, mean heads) ===========
    run_gemm(rstB, W2, feat, packA, packB, true, N, F01, F2, stream);
    run_gemm(rstB, Wres2, res2, packA, packB, false, N, F01, F2, stream); // reuse packed A
    run_edge_phase(feat, src, dst, attn2, logit, nmax, denom, rstA, N, E, HO, C, stream);
    final_mean_kernel<<<dim3(cdiv((long)N * C, 256)), dim3(256), 0, stream>>>(
        rstA, res2, (float*)d_out, N, HO, C);
}
